// SharedEncoder_13400297963521
// MI455X (gfx1250) — matrix operand order repaired and checked by reading, not compiled, not run on hardware
//
#include <hip/hip_runtime.h>
#include <hip/hip_bf16.h>
#include <stdint.h>

// ---------------------------------------------------------------------------
// LSTM encoder for MI455X (gfx1250, wave32, WMMA).
//
// Roofline: 103 GFLOP total but only ~40MB HBM traffic -> the 512-step serial
// recurrence dominates; everything parallel is hoisted out (bf16 embedding
// gather, bf16 weight swizzle), then ONE persistent kernel runs all 512 steps.
//
// Key CDNA5 feature used here: 1024 VGPRs/wave (S_SET_VGPR_MSB addressing).
// Each wave owns a 16x16 output tile for all 4 gates; its weight slice is
// 4 gates x 24 K-steps x 32B = 768 VGPRs -> the weights are REGISTER-
// STATIONARY for the entire sequence (loaded once), so the steady-state loop
// is just A-fragment loads from L2 + 96 back-to-back v_wmma_f32_16x16x32_bf16
// per step. c/h state also lives in VGPRs. Cross-workgroup h exchange (64KB
// bf16) goes through L2; the 4 batch-row groups are independent, so each row
// group has its own 16-WG atomic barrier (separate cache lines).
// Fragment layouts per CDNA5 ISA 7.12.2.
// ---------------------------------------------------------------------------

typedef __attribute__((ext_vector_type(16))) __bf16 v16bf;
typedef __attribute__((ext_vector_type(8)))  __bf16 v8bf;
typedef __attribute__((ext_vector_type(8)))  float  v8f;

#define B_    64
#define T_    512
#define D_    256
#define H_    512
#define G4H   2048
#define KTOT  768            // D + H
#define NKT   24             // KTOT / 32 (WMMA K-steps)
#define NKT_X 8              // D / 32   (x-part K-steps)
#define CG_   16             // column groups (32 h-cols each)
#define RG_   4              // row groups (16 batch rows each)
#define NWG   (CG_*RG_)      // 64 workgroups
#define WVS   2              // waves per WG (each owns 16 h-cols x 4 gates)
#define THR   (WVS*32)

#define WG_W_ELEMS (WVS*4*NKT*512)        // 98304 bf16 per col-group

// workspace layout (bytes); total ~20MB
#define WSWZ_OFF   ((size_t)0)
#define WSWZ_BYTES ((size_t)CG_*WG_W_ELEMS*2)   // 3,145,728
#define XBF_OFF    (WSWZ_OFF + WSWZ_BYTES)
#define XBF_BYTES  ((size_t)B_*T_*D_*2)         // 16,777,216
#define HBF_OFF    (XBF_OFF + XBF_BYTES)
#define HBF_BYTES  ((size_t)B_*H_*2)            // 65,536
#define CNT_OFF    (HBF_OFF + HBF_BYTES)
#define CNT_STRIDE 64                           // uints: 256B per row group
#define CNT_BYTES  ((size_t)RG_*CNT_STRIDE*4)   // 1024

// ---------------------------------------------------------------------------
// Prep 1: swizzle fp32 kernel[(D+H),4H] (row-major) into bf16 WMMA-B fragment
// order: [cg][wave][gate][kt][lane][16 K in table order]. B-matrix lane layout
// (ISA 7.12.2, 16-bit B 32x16): lane l -> column N = l&15, K = (l>>4)*8..+7 and (l>>4)*8+16..+23.
// ---------------------------------------------------------------------------
__global__ void prep_weights(const float* __restrict__ kern,
                             __bf16* __restrict__ wswz) {
  int tid = blockIdx.x * blockDim.x + threadIdx.x;       // 0 .. 98303
  if (tid >= CG_ * WVS * 4 * NKT * 32) return;
  int l  = tid & 31;
  int kt = (tid >> 5) % NKT;
  int g  = ((tid >> 5) / NKT) % 4;
  int wv = ((tid >> 5) / (NKT * 4)) % WVS;
  int cg = tid / (32 * NKT * 4 * WVS);
  int n  = g * H_ + cg * 32 + wv * 16 + (l & 15);        // gate-space column
  int kb = kt * 32 + (l >> 4) * 8;                       // K base for this lane
  __bf16* dst = wswz + (size_t)tid * 16;
#pragma unroll
  for (int j = 0; j < 16; ++j)
    dst[j] = (__bf16)kern[(size_t)(kb + (j & 7) + (j >> 3) * 16) * G4H + n];
}

// ---------------------------------------------------------------------------
// Prep 2: embedding gather fp32 -> bf16, layout Xbf[(t*B + b)*D + k] so the
// recurrent kernel's A-fragment loads are contiguous per (t, batch-row).
// ---------------------------------------------------------------------------
__global__ void prep_x(const int* __restrict__ idx,
                       const float* __restrict__ emb,
                       __bf16* __restrict__ xbf) {
  size_t tid = (size_t)blockIdx.x * blockDim.x + threadIdx.x;
  if (tid >= (size_t)B_ * T_ * D_) return;
  int    k  = (int)(tid & (D_ - 1));
  size_t bt = tid >> 8;                 // = t*B + b
  int    b  = (int)(bt & (B_ - 1));
  int    t  = (int)(bt >> 6);
  int    w  = idx[b * T_ + t];
  xbf[tid] = (__bf16)emb[(size_t)w * D_ + k];
}

// ---------------------------------------------------------------------------
// Persistent recurrent kernel.
// ---------------------------------------------------------------------------
__device__ __forceinline__ float sigf(float x) {
  return 1.0f / (1.0f + __expf(-x));
}
__device__ __forceinline__ float tanh_f(float x) {
  return 1.0f - 2.0f / (__expf(2.0f * x) + 1.0f);
}
__device__ __forceinline__ v8f vzero8() {
  v8f z = {0.f, 0.f, 0.f, 0.f, 0.f, 0.f, 0.f, 0.f};
  return z;
}

extern "C" __global__ __launch_bounds__(THR, 1)
void lstm_wmma(const __bf16* __restrict__ xbf,
               __bf16* __restrict__ hbf,
               const __bf16* __restrict__ wswz,
               const int* __restrict__ num_words,
               const float* __restrict__ bias,
               float* __restrict__ out,
               unsigned int* __restrict__ cnt) {
  const int tid   = threadIdx.x;
  const int wv    = tid >> 5;                      // wave in WG (0..1)
  const int lane  = tid & 31;
  const int l15   = lane & 15;
  const int lh    = lane >> 4;                     // lane half
  const int rg    = blockIdx.x & (RG_ - 1);        // batch-row group
  const int cg    = blockIdx.x / RG_;              // h-column group
  const int r0    = rg * 16;
  const int mycol = cg * 32 + wv * 16 + l15;       // owned h-column (C/D: N=l&15)
  unsigned int* mycnt = cnt + rg * CNT_STRIDE;     // per-row-group barrier

  // -------------------------------------------------------------------------
  // Load this wave's weight slice into registers ONCE: 96 x v16bf = 768 VGPRs,
  // stationary across all 512 timesteps (gfx1250: 1024 VGPRs/wave).
  // bfr[g*NKT + kt] is the B operand for gate g, K-step kt.
  // -------------------------------------------------------------------------
  const __bf16* wgb = wswz + (size_t)cg * WG_W_ELEMS +
                      (size_t)wv * (4 * NKT * 512) + (size_t)lane * 16;
  v16bf bfr[4 * NKT];
#pragma unroll
  for (int i = 0; i < 4 * NKT; ++i)
    bfr[i] = *(const v16bf*)(wgb + (size_t)i * 512);

  // Per-lane constants. C/D layout: VGPR e holds row M = e + lh*8, col l&15.
  const float bi  = bias[mycol];
  const float bj  = bias[H_ + mycol];
  const float bf1 = bias[2 * H_ + mycol] + 1.0f;   // TF forget bias folded in
  const float bo  = bias[3 * H_ + mycol];
  int nw[8];
#pragma unroll
  for (int e = 0; e < 8; ++e) nw[e] = num_words[r0 + lh * 8 + e];

  float c_r[8], h_r[8];
#pragma unroll
  for (int e = 0; e < 8; ++e) { c_r[e] = 0.0f; h_r[e] = 0.0f; }

  // A-fragment layout (16-bit A 16x32): lane l -> row M = l&15;
  // K = (l>>4)*8 .. +7 in VGPRs0-3 and +16 more in VGPRs4-7.
  const int arow = r0 + l15;                       // batch row this lane loads
  const __bf16* hrow = hbf + (size_t)arow * H_ + lh * 8;

  for (int t = 0; t < T_; ++t) {
    const __bf16* xrow = xbf + ((size_t)t * B_ + arow) * D_ + lh * 8;
    if (t + 1 < T_)                                 // pull next step's x slice
      __builtin_prefetch((const void*)(xrow + (size_t)B_ * D_), 0, 1);

    v8f a0 = vzero8(), a1 = vzero8(), a2 = vzero8(), a3 = vzero8();

#pragma unroll
    for (int kt = 0; kt < NKT; ++kt) {
      const __bf16* ap = (kt < NKT_X) ? (xrow + kt * 32)
                                      : (hrow + (kt - NKT_X) * 32);
      v8bf lo = *(const v8bf*)ap;                   // K = kbase .. +7
      v8bf hi = *(const v8bf*)(ap + 16);           // K = kbase+16 .. +23
      v16bf af;
#pragma unroll
      for (int j = 0; j < 8; ++j) { af[j] = lo[j]; af[8 + j] = hi[j]; }

      a0 = __builtin_amdgcn_wmma_f32_16x16x32_bf16(false, af, false,
                                                   bfr[0 * NKT + kt],
                                                   (short)0, a0, false, false);
      a1 = __builtin_amdgcn_wmma_f32_16x16x32_bf16(false, af, false,
                                                   bfr[1 * NKT + kt],
                                                   (short)0, a1, false, false);
      a2 = __builtin_amdgcn_wmma_f32_16x16x32_bf16(false, af, false,
                                                   bfr[2 * NKT + kt],
                                                   (short)0, a2, false, false);
      a3 = __builtin_amdgcn_wmma_f32_16x16x32_bf16(false, af, false,
                                                   bfr[3 * NKT + kt],
                                                   (short)0, a3, false, false);
    }

    // Gate math; c/h state lives in registers, masked per TF sequence_length.
#pragma unroll
    for (int e = 0; e < 8; ++e) {
      float cn = sigf(a2[e] + bf1) * c_r[e] + sigf(a0[e] + bi) * tanh_f(a1[e] + bj);
      float hn = sigf(a3[e] + bo) * tanh_f(cn);
      if (t < nw[e]) { c_r[e] = cn; h_r[e] = hn; }
      hbf[(size_t)(r0 + lh * 8 + e) * H_ + mycol] = (__bf16)h_r[e];
    }

    // Row-group barrier: only the 16 WGs sharing batch rows exchange h.
    if (t + 1 < T_) {
      __threadfence();
      __syncthreads();
      if (tid == 0) {
        __hip_atomic_fetch_add(mycnt, 1u, __ATOMIC_ACQ_REL,
                               __HIP_MEMORY_SCOPE_AGENT);
        const unsigned target = (unsigned)CG_ * (unsigned)(t + 1);
        while (__hip_atomic_load(mycnt, __ATOMIC_ACQUIRE,
                                 __HIP_MEMORY_SCOPE_AGENT) < target)
          __builtin_amdgcn_s_sleep(1);
      }
      __syncthreads();
      __threadfence();
    }
  }

  // Final state: out = stack([c, h]) as [2, B, H] fp32.
#pragma unroll
  for (int e = 0; e < 8; ++e) {
    const size_t row = (size_t)(r0 + lh * 8 + e);
    out[row * H_ + mycol]                   = c_r[e];
    out[(size_t)B_ * H_ + row * H_ + mycol] = h_r[e];
  }
}

// ---------------------------------------------------------------------------
extern "C" void kernel_launch(void* const* d_in, const int* in_sizes, int n_in,
                              void* d_out, int out_size, void* d_ws,
                              size_t ws_size, hipStream_t stream) {
  const int*   widx = (const int*)d_in[0];    // word_indices [B, T]
  const int*   nwp  = (const int*)d_in[1];    // num_words    [B]
  const float* emb  = (const float*)d_in[2];  // W_emb        [V, D]
  const float* kern = (const float*)d_in[3];  // kernel       [D+H, 4H]
  const float* bias = (const float*)d_in[4];  // bias         [4H]
  float* out = (float*)d_out;

  char* ws = (char*)d_ws;
  __bf16*   wswz = (__bf16*)(ws + WSWZ_OFF);
  __bf16*   xbf  = (__bf16*)(ws + XBF_OFF);
  __bf16*   hbf  = (__bf16*)(ws + HBF_OFF);
  unsigned* cnt  = (unsigned*)(ws + CNT_OFF);

  // zero h0 and the barrier counters (re-done every call: deterministic).
  hipMemsetAsync(ws + HBF_OFF, 0, HBF_BYTES + CNT_BYTES, stream);

  prep_weights<<<(CG_ * WVS * 4 * NKT * 32 + 255) / 256, 256, 0, stream>>>(
      kern, wswz);
  {
    size_t n = (size_t)B_ * T_ * D_;
    prep_x<<<(unsigned)((n + 255) / 256), 256, 0, stream>>>(widx, emb, xbf);
  }

  lstm_wmma<<<NWG, THR, 0, stream>>>(xbf, hbf, wswz, nwp, bias, out, cnt);
}
